// RecommendationModel_50689204027711
// MI455X (gfx1250) — compile-verified, matrix operand order checked by reading, not run on hardware
//
#include <hip/hip_runtime.h>
#include <math.h>

#define TPB 256
#define BM 128
#define BN 64
#define BK 32

#define NNODE 30000
#define MP    30080            // NNODE padded to multiple of BM
#define HID   256
#define EBUYS 200000

typedef _Float16 v16h __attribute__((ext_vector_type(16)));
typedef _Float16 v8h  __attribute__((ext_vector_type(8)));
typedef float    v8f  __attribute__((ext_vector_type(8)));

union HalfFrag { v16h v; v8h h[2]; };

// -------------------------------------------------------------------------
// Tiled GEMM: C = A(f16)[M x K] * B(f16)[K x N] (+bias), f32 acc via
// v_wmma_f32_16x16x32_f16. B is supplied TRANSPOSED: Bt[N x K], ldbt.
// A and Bt tiles are staged to LDS with two (A) / one (Bt) b128 load + b128 store
// per thread (no narrow transposing DS stores in the hot loop).
// M multiple of BM, N of BN, K of BK. Optional f32 out (row-guarded) + f16 out.
// -------------------------------------------------------------------------
__global__ __launch_bounds__(TPB)
void gemm_f16(const _Float16* __restrict__ A, int lda,
              const _Float16* __restrict__ Bt, int ldbt,
              const float* __restrict__ bias,
              float* __restrict__ Cf, int ldcf, int Mf,
              _Float16* __restrict__ Ch, int ldch,
              int K)
{
    __shared__ _Float16 sA[BM * BK];   // [m][k]
    __shared__ _Float16 sB[BN * BK];   // [n][k]

    const int tid  = threadIdx.x;
    const int lane = tid & 31;
    const int w    = tid >> 5;   // 8 waves
    const int wm   = w & 3;      // 4 along M
    const int wn   = w >> 2;     // 2 along N
    const long long bm = (long long)blockIdx.y * BM;
    const long long bn = (long long)blockIdx.x * BN;

    v8f acc[2][2];
#pragma unroll
    for (int i = 0; i < 2; ++i)
#pragma unroll
        for (int j = 0; j < 2; ++j)
#pragma unroll
            for (int t = 0; t < 8; ++t) acc[i][j][t] = 0.f;

    const int ar = tid >> 1;           // A tile: row 0..127
    const int ak = (tid & 1) * 16;     //          k 0 / 16
    const int bnr = tid >> 2;          // B tile: n row 0..63
    const int bkp = (tid & 3) * 8;     //          k 0/8/16/24
    const int koff = (lane >> 4) * 8;  // wmma fragment k-offset per half-wave
    const int fr   = lane & 15;        // fragment row (A) / col (B)

    for (int kk = 0; kk < K; kk += BK) {
        const _Float16* ga = A + (bm + ar) * (long long)lda + kk + ak;
        *(v8h*)&sA[ar * BK + ak]     = *(const v8h*)(ga);
        *(v8h*)&sA[ar * BK + ak + 8] = *(const v8h*)(ga + 8);
        if (kk + BK < K) __builtin_prefetch(ga + BK, 0, 1);

        const _Float16* gb = Bt + (bn + bnr) * (long long)ldbt + kk + bkp;
        *(v8h*)&sB[bnr * BK + bkp] = *(const v8h*)(gb);
        __syncthreads();

        HalfFrag af[2], bf[2];
#pragma unroll
        for (int i = 0; i < 2; ++i) {
            int arow = wm * 32 + i * 16 + fr;
            af[i].h[0] = *(const v8h*)&sA[arow * BK + koff];
            af[i].h[1] = *(const v8h*)&sA[arow * BK + 16 + koff];
            int bcol = wn * 32 + i * 16 + fr;
            bf[i].h[0] = *(const v8h*)&sB[bcol * BK + koff];
            bf[i].h[1] = *(const v8h*)&sB[bcol * BK + 16 + koff];
        }
#pragma unroll
        for (int mi = 0; mi < 2; ++mi)
#pragma unroll
            for (int ni = 0; ni < 2; ++ni)
                acc[mi][ni] = __builtin_amdgcn_wmma_f32_16x16x32_f16(
                    false, af[mi].v, false, bf[ni].v,
                    (short)0, acc[mi][ni], false, false);
        __syncthreads();
    }

    const int r8 = (lane >> 4) * 8;
#pragma unroll
    for (int mi = 0; mi < 2; ++mi) {
#pragma unroll
        for (int ni = 0; ni < 2; ++ni) {
            long long col = bn + wn * 32 + ni * 16 + fr;
            float bb = bias ? bias[col] : 0.f;
#pragma unroll
            for (int j = 0; j < 8; ++j) {
                long long row = bm + wm * 32 + mi * 16 + r8 + j;
                float c = acc[mi][ni][j] + bb;
                if (Cf && row < Mf) Cf[row * (long long)ldcf + col] = c;
                if (Ch) Ch[row * (long long)ldch + col] = (_Float16)c;
            }
        }
    }
}

// ----------------------------- helpers -----------------------------------
__global__ void k_fill(float* p, float v, long long n) {
    long long id = (long long)blockIdx.x * TPB + threadIdx.x;
    if (id < n) p[id] = v;
}

// f32 [M x K] -> f16 [Mp x Kp], zero padded (activations)
__global__ void k_conv(const float* __restrict__ src, int M, int K,
                       _Float16* __restrict__ dst, int Mp, int Kp) {
    long long id = (long long)blockIdx.x * TPB + threadIdx.x;
    if (id >= (long long)Mp * Kp) return;
    int m = (int)(id / Kp), k = (int)(id % Kp);
    dst[id] = (m < M && k < K) ? (_Float16)src[(long long)m * K + k] : (_Float16)0.f;
}

// Per-block transpose+convert: src = nblk blocks of f32 [K x N] (row-major),
// dst = nblk blocks of f16 [N x Kp] (row-major, k-padded with zeros).
__global__ void k_convT(const float* __restrict__ src, int nblk, int K, int N, int Kp,
                        _Float16* __restrict__ dst) {
    long long id = (long long)blockIdx.x * TPB + threadIdx.x;
    if (id >= (long long)nblk * N * Kp) return;
    long long per = (long long)N * Kp;
    int b = (int)(id / per);
    long long r = id % per;
    int n = (int)(r / Kp), k = (int)(r % Kp);
    dst[id] = (k < K) ? (_Float16)src[((long long)b * K + k) * N + n] : (_Float16)0.f;
}

// gather rows of emb by idx -> f16 [Mp x K], zero padded
__global__ void k_gather(const float* __restrict__ emb, const int* __restrict__ idx,
                         _Float16* __restrict__ dst, int M, int Mp, int K) {
    long long id = (long long)blockIdx.x * TPB + threadIdx.x;
    if (id >= (long long)Mp * K) return;
    int m = (int)(id / K), k = (int)(id % K);
    dst[id] = (m < M) ? (_Float16)emb[(long long)idx[m] * K + k] : (_Float16)0.f;
}

// column partial sums (sum, sumsq) of X[M x 256] -> acc[512] via atomics
__global__ void k_bn_partial(const float* __restrict__ X, int M, float* __restrict__ acc) {
    int c = threadIdx.x;  // 256 columns
    float s = 0.f, s2 = 0.f;
    for (int r = blockIdx.x; r < M; r += gridDim.x) {
        float v = X[(long long)r * HID + c];
        s += v; s2 += v * v;
    }
    atomicAdd(acc + c, s);
    atomicAdd(acc + HID + c, s2);
}

// y = relu(BN(x)); write f32 (rows<M) and f16 (Mp rows, pad=0)
__global__ void k_bn_apply(const float* __restrict__ X, const float* __restrict__ acc,
                           const float* __restrict__ g, const float* __restrict__ b,
                           float invM, float* __restrict__ outF, _Float16* __restrict__ outH,
                           int M, int Mp) {
    long long id = (long long)blockIdx.x * TPB + threadIdx.x;
    if (id >= (long long)Mp * HID) return;
    int m = (int)(id >> 8), c = (int)(id & 255);
    float mu = acc[c] * invM;
    float var = acc[HID + c] * invM - mu * mu;
    float y = 0.f;
    if (m < M) {
        float v = (X[id] - mu) * rsqrtf(var + 1e-5f) * g[c] + b[c];
        y = fmaxf(v, 0.f);
        outF[(long long)m * HID + c] = y;
    }
    if (outH) outH[id] = (_Float16)y;
}

__device__ inline void atomicMaxF(float* addr, float val) {
    int* ia = (int*)addr;
    int cur = __float_as_int(*addr);
    while (__int_as_float(cur) < val) {
        int prev = atomicCAS(ia, cur, __float_as_int(val));
        if (prev == cur) break;
        cur = prev;
    }
}

__device__ inline float dot64h(const _Float16* __restrict__ a, const _Float16* __restrict__ b) {
    float s = 0.f;
#pragma unroll
    for (int j = 0; j < 8; ++j) {
        v8h x = *(const v8h*)(a + 8 * j);
        v8h y = *(const v8h*)(b + 8 * j);
#pragma unroll
        for (int t = 0; t < 8; ++t) s += (float)x[t] * (float)y[t];
    }
    return s;
}

// assoc (identity) edges e=0..2 into items: logits + direct max init
__global__ void k_att_assoc_logits(const _Float16* __restrict__ kqv_item,
                                   const _Float16* __restrict__ ke0,
                                   const _Float16* __restrict__ ke1,
                                   const _Float16* __restrict__ ke2,
                                   const float* __restrict__ prl, float scale,
                                   float* la0, float* la1, float* la2, float* m_item) {
    long long id = (long long)blockIdx.x * TPB + threadIdx.x;
    if (id >= (long long)NNODE * 4) return;
    int i = (int)(id >> 2), h = (int)(id & 3);
    const _Float16* q = kqv_item + (long long)i * 768 + 256 + h * 64;
    float l0 = dot64h(q, ke0 + (long long)i * HID + h * 64) * prl[0 * 4 + h] * scale;
    float l1 = dot64h(q, ke1 + (long long)i * HID + h * 64) * prl[1 * 4 + h] * scale;
    float l2 = dot64h(q, ke2 + (long long)i * HID + h * 64) * prl[2 * 4 + h] * scale;
    la0[id] = l0; la1[id] = l1; la2[id] = l2;
    m_item[id] = fmaxf(l0, fmaxf(l1, l2));
}

// buys edges: e=3 (user->item) and e=4 (item->user)
__global__ void k_att_buys_logits(const int* __restrict__ bsrc, const int* __restrict__ bdst,
                                  const _Float16* __restrict__ kqv_item,
                                  const _Float16* __restrict__ kqv_user,
                                  const _Float16* __restrict__ ke3,
                                  const _Float16* __restrict__ ke4,
                                  const float* __restrict__ prl, float scale,
                                  float* lb3, float* lb4, float* m_item, float* m_user) {
    long long id = (long long)blockIdx.x * TPB + threadIdx.x;
    if (id >= (long long)EBUYS * 4) return;
    int e = (int)(id >> 2), h = (int)(id & 3);
    int us = bsrc[e], it = bdst[e];
    const _Float16* qi = kqv_item + (long long)it * 768 + 256 + h * 64;
    float l3 = dot64h(qi, ke3 + (long long)us * HID + h * 64) * prl[3 * 4 + h] * scale;
    lb3[id] = l3;
    atomicMaxF(m_item + (long long)it * 4 + h, l3);
    const _Float16* qu = kqv_user + (long long)us * 768 + 256 + h * 64;
    float l4 = dot64h(qu, ke4 + (long long)it * HID + h * 64) * prl[4 * 4 + h] * scale;
    lb4[id] = l4;
    atomicMaxF(m_user + (long long)us * 4 + h, l4);
}

__global__ void k_att_assoc_exp(float* la0, float* la1, float* la2,
                                const float* __restrict__ m_item, float* s_item) {
    long long id = (long long)blockIdx.x * TPB + threadIdx.x;
    if (id >= (long long)NNODE * 4) return;
    float m = m_item[id];
    float e0 = expf(la0[id] - m), e1 = expf(la1[id] - m), e2 = expf(la2[id] - m);
    la0[id] = e0; la1[id] = e1; la2[id] = e2;
    s_item[id] = e0 + e1 + e2;   // direct init; buys adds on top
}

__global__ void k_att_buys_exp(float* lb3, float* lb4,
                               const int* __restrict__ bsrc, const int* __restrict__ bdst,
                               const float* __restrict__ m_item, const float* __restrict__ m_user,
                               float* s_item, float* s_user) {
    long long id = (long long)blockIdx.x * TPB + threadIdx.x;
    if (id >= (long long)EBUYS * 4) return;
    int e = (int)(id >> 2), h = (int)(id & 3);
    int us = bsrc[e], it = bdst[e];
    float e3 = expf(lb3[id] - m_item[(long long)it * 4 + h]);
    lb3[id] = e3;
    atomicAdd(s_item + (long long)it * 4 + h, e3);
    float e4 = expf(lb4[id] - m_user[(long long)us * 4 + h]);
    lb4[id] = e4;
    atomicAdd(s_user + (long long)us * 4 + h, e4);
}

__global__ void k_att_assoc_agg(const float* __restrict__ la0, const float* __restrict__ la1,
                                const float* __restrict__ la2, const float* __restrict__ s_item,
                                const _Float16* __restrict__ ve0, const _Float16* __restrict__ ve1,
                                const _Float16* __restrict__ ve2, float* __restrict__ agg_item) {
    long long id = (long long)blockIdx.x * TPB + threadIdx.x;
    if (id >= (long long)NNODE * HID) return;
    int i = (int)(id >> 8), c = (int)(id & 255), h = c >> 6;
    float inv = 1.f / (s_item[(long long)i * 4 + h] + 1e-16f);
    float a0 = la0[(long long)i * 4 + h] * inv;
    float a1 = la1[(long long)i * 4 + h] * inv;
    float a2 = la2[(long long)i * 4 + h] * inv;
    agg_item[id] = a0 * (float)ve0[id] + a1 * (float)ve1[id] + a2 * (float)ve2[id];
}

__global__ void k_att_buys_agg(const float* __restrict__ lb3, const float* __restrict__ lb4,
                               const int* __restrict__ bsrc, const int* __restrict__ bdst,
                               const float* __restrict__ s_item, const float* __restrict__ s_user,
                               const _Float16* __restrict__ ve3, const _Float16* __restrict__ ve4,
                               float* agg_item, float* agg_user) {
    long long id = (long long)blockIdx.x * TPB + threadIdx.x;
    if (id >= (long long)EBUYS * HID) return;
    int e = (int)(id >> 8), c = (int)(id & 255), h = c >> 6;
    int us = bsrc[e], it = bdst[e];
    float a3 = lb3[(long long)e * 4 + h] / (s_item[(long long)it * 4 + h] + 1e-16f);
    atomicAdd(agg_item + (long long)it * HID + c, a3 * (float)ve3[(long long)us * HID + c]);
    float a4 = lb4[(long long)e * 4 + h] / (s_user[(long long)us * 4 + h] + 1e-16f);
    atomicAdd(agg_user + (long long)us * HID + c, a4 * (float)ve4[(long long)it * HID + c]);
}

__global__ void k_gelu_f16(const float* __restrict__ x, _Float16* __restrict__ y, int M, int Mp) {
    long long id = (long long)blockIdx.x * TPB + threadIdx.x;
    if (id >= (long long)Mp * HID) return;
    int m = (int)(id >> 8);
    float v = 0.f;
    if (m < M) {
        float t = x[id];
        v = 0.5f * t * (1.f + erff(t * 0.70710678118654752f));
    }
    y[id] = (_Float16)v;
}

__global__ void k_fuse(float* o, const float* __restrict__ x, const float* __restrict__ skip,
                       int idx, long long n) {
    long long id = (long long)blockIdx.x * TPB + threadIdx.x;
    if (id >= n) return;
    float sg = 1.f / (1.f + expf(-skip[idx]));
    float oo = o[id], xx = x[id];
    o[id] = sg * oo + (1.f - sg) * xx + xx;
}

// -------------------------------------------------------------------------
extern "C" void kernel_launch(void* const* d_in, const int* in_sizes, int n_in,
                              void* d_out_, int out_size, void* d_ws, size_t ws_size,
                              hipStream_t stream) {
    (void)in_sizes; (void)n_in; (void)out_size; (void)ws_size;
    const int*   user_id    = (const int*)d_in[0];
    const int*   item_id    = (const int*)d_in[1];
    const int*   bsrc       = (const int*)d_in[2];
    const int*   bdst       = (const int*)d_in[3];
    const float* image_feat = (const float*)d_in[4];
    const float* taste_feat = (const float*)d_in[5];
    const float* intent_feat= (const float*)d_in[6];
    const float* user_emb   = (const float*)d_in[7];
    const float* user_W     = (const float*)d_in[8];
    const float* user_b     = (const float*)d_in[9];
    const float* user_bn_g  = (const float*)d_in[10];
    const float* user_bn_b  = (const float*)d_in[11];
    const float* item_emb   = (const float*)d_in[12];
    const float* item_W     = (const float*)d_in[13];
    const float* item_b     = (const float*)d_in[14];
    const float* item_bn_g  = (const float*)d_in[15];
    const float* item_bn_b  = (const float*)d_in[16];
    const float* img_U      = (const float*)d_in[17];
    const float* img_V      = (const float*)d_in[18];
    const float* img_b      = (const float*)d_in[19];
    const float* taste_W    = (const float*)d_in[20];
    const float* taste_b    = (const float*)d_in[21];
    const float* intent_W   = (const float*)d_in[22];
    const float* intent_b   = (const float*)d_in[23];
    const float* kqv_W      = (const float*)d_in[24];
    const float* kqv_b      = (const float*)d_in[25];
    const float* out_W      = (const float*)d_in[26];
    const float* out_b      = (const float*)d_in[27];
    const float* k_rel      = (const float*)d_in[28];
    const float* v_rel      = (const float*)d_in[29];
    const float* p_rel      = (const float*)d_in[30];
    const float* skip       = (const float*)d_in[31];
    const float* fbn_g      = (const float*)d_in[32];
    const float* fbn_b      = (const float*)d_in[33];
    float* dout = (float*)d_out_;

    const long long sect = (long long)NNODE * HID;

    // ---- workspace bump allocator ----
    char* ws = (char*)d_ws;
    size_t off = 0;
    auto alloc = [&](size_t bytes) -> char* {
        char* p = ws + off;
        off = (off + bytes + 255) & ~(size_t)255;
        return p;
    };
    const size_t F16 = sizeof(_Float16), F32 = sizeof(float);

    _Float16 *hx[5], *hkqv[5], *hke[5], *hve[5];
    for (int t = 0; t < 5; ++t) hx[t]   = (_Float16*)alloc((size_t)MP * HID * F16);
    for (int t = 0; t < 5; ++t) hkqv[t] = (_Float16*)alloc((size_t)MP * 768 * F16);
    for (int e = 0; e < 5; ++e) hke[e]  = (_Float16*)alloc((size_t)MP * HID * F16);
    for (int e = 0; e < 5; ++e) hve[e]  = (_Float16*)alloc((size_t)MP * HID * F16);
    _Float16* h_in   = (_Float16*)alloc((size_t)MP * 768 * F16);
    _Float16* h_mid  = (_Float16*)alloc((size_t)MP * 64 * F16);
    _Float16* h_gelu = (_Float16*)alloc((size_t)MP * HID * F16);

    // all weights stored TRANSPOSED as f16: [N x Kp] per block
    _Float16* wT_user  = (_Float16*)alloc((size_t)256 * 64 * F16);
    _Float16* wT_item  = (_Float16*)alloc((size_t)256 * 64 * F16);
    _Float16* wT_imgU  = (_Float16*)alloc((size_t)64 * 512 * F16);
    _Float16* wT_imgV  = (_Float16*)alloc((size_t)256 * 64 * F16);
    _Float16* wT_taste = (_Float16*)alloc((size_t)256 * 768 * F16);
    _Float16* wT_intent= (_Float16*)alloc((size_t)256 * 32 * F16);
    _Float16* wT_kqv   = (_Float16*)alloc((size_t)10 * 768 * 256 * F16);
    _Float16* wT_out   = (_Float16*)alloc((size_t)10 * 256 * 256 * F16);
    _Float16* wT_krel  = (_Float16*)alloc((size_t)40 * 64 * 64 * F16);
    _Float16* wT_vrel  = (_Float16*)alloc((size_t)40 * 64 * 64 * F16);

    float* o_tmp    = (float*)alloc((size_t)MP * HID * F32);
    float* agg_item = (float*)alloc((size_t)MP * HID * F32);
    float* agg_user = (float*)alloc((size_t)MP * HID * F32);
    float* m_item = (float*)alloc((size_t)NNODE * 4 * F32);
    float* s_item = (float*)alloc((size_t)NNODE * 4 * F32);
    float* m_user = (float*)alloc((size_t)NNODE * 4 * F32);
    float* s_user = (float*)alloc((size_t)NNODE * 4 * F32);
    float* la0 = (float*)alloc((size_t)NNODE * 4 * F32);
    float* la1 = (float*)alloc((size_t)NNODE * 4 * F32);
    float* la2 = (float*)alloc((size_t)NNODE * 4 * F32);
    float* lb3 = (float*)alloc((size_t)EBUYS * 4 * F32);
    float* lb4 = (float*)alloc((size_t)EBUYS * 4 * F32);
    float* bnacc = (float*)alloc((size_t)512 * F32);

    auto nb = [](long long n) { return (unsigned)((n + TPB - 1) / TPB); };
    auto gemm = [&](const _Float16* A, int lda, const _Float16* Bt, int ldbt,
                    const float* bias, float* Cf, int ldcf, int Mf,
                    _Float16* Ch, int ldch, int M, int N, int K) {
        dim3 grid((unsigned)(N / BN), (unsigned)(M / BM));
        gemm_f16<<<grid, TPB, 0, stream>>>(A, lda, Bt, ldbt, bias, Cf, ldcf, Mf, Ch, ldch, K);
    };
    auto conv = [&](const float* s, int M, int K, _Float16* d, int Mp2, int Kp) {
        k_conv<<<nb((long long)Mp2 * Kp), TPB, 0, stream>>>(s, M, K, d, Mp2, Kp);
    };
    auto convT = [&](const float* s, int nblk, int K, int N, int Kp, _Float16* d) {
        k_convT<<<nb((long long)nblk * N * Kp), TPB, 0, stream>>>(s, nblk, K, N, Kp, d);
    };
    auto fill = [&](float* p, float v, long long n) {
        k_fill<<<nb(n), TPB, 0, stream>>>(p, v, n);
    };
    auto bn = [&](const float* X, const float* g, const float* b, float* outF, _Float16* outH) {
        fill(bnacc, 0.f, 512);
        k_bn_partial<<<256, 256, 0, stream>>>(X, NNODE, bnacc);
        k_bn_apply<<<nb((long long)MP * HID), TPB, 0, stream>>>(
            X, bnacc, g, b, 1.f / (float)NNODE, outF, outH, NNODE, MP);
    };

    // ---- convert (and transpose) weights to f16 ----
    convT(user_W, 1, 64, 256, 64, wT_user);
    convT(item_W, 1, 64, 256, 64, wT_item);
    convT(img_U, 1, 512, 64, 512, wT_imgU);
    convT(img_V, 1, 64, 256, 64, wT_imgV);
    convT(taste_W, 1, 768, 256, 768, wT_taste);
    convT(intent_W, 1, 20, 256, 32, wT_intent);          // K padded 20->32
    convT(kqv_W, 10, 256, 768, 256, wT_kqv);
    convT(out_W, 10, 256, 256, 256, wT_out);
    convT(k_rel, 40, 64, 64, 64, wT_krel);
    convT(v_rel, 40, 64, 64, 64, wT_vrel);

    // ---- encoders ----
    // user
    k_gather<<<nb((long long)MP * 64), TPB, 0, stream>>>(user_emb, user_id, h_in, NNODE, MP, 64);
    gemm(h_in, 64, wT_user, 64, user_b, o_tmp, HID, NNODE, nullptr, 0, MP, 256, 64);
    bn(o_tmp, user_bn_g, user_bn_b, dout + 0 * sect, hx[0]);
    // item
    k_gather<<<nb((long long)MP * 64), TPB, 0, stream>>>(item_emb, item_id, h_in, NNODE, MP, 64);
    gemm(h_in, 64, wT_item, 64, item_b, o_tmp, HID, NNODE, nullptr, 0, MP, 256, 64);
    bn(o_tmp, item_bn_g, item_bn_b, dout + 1 * sect, hx[1]);
    // taste
    conv(taste_feat, NNODE, 768, h_in, MP, 768);
    gemm(h_in, 768, wT_taste, 768, taste_b, dout + 2 * sect, HID, NNODE, hx[2], HID, MP, 256, 768);
    // intention (K padded to 32)
    conv(intent_feat, NNODE, 20, h_in, MP, 32);
    gemm(h_in, 32, wT_intent, 32, intent_b, dout + 3 * sect, HID, NNODE, hx[3], HID, MP, 256, 32);
    // image low-rank
    conv(image_feat, NNODE, 512, h_in, MP, 512);
    gemm(h_in, 512, wT_imgU, 512, nullptr, nullptr, 0, 0, h_mid, 64, MP, 64, 512);
    gemm(h_mid, 64, wT_imgV, 64, img_b, dout + 4 * sect, HID, NNODE, hx[4], HID, MP, 256, 64);

    // ---- HGT layers ----
    const int srcOf[5] = {2, 3, 4, 0, 1};
    const float scale = 0.125f;  // 1/sqrt(64)
    for (int l = 0; l < 2; ++l) {
        // kqv projections per node type
        for (int t = 0; t < 5; ++t)
            gemm(hx[t], 256, wT_kqv + (size_t)(l * 5 + t) * 768 * 256, 256,
                 kqv_b + (size_t)(l * 5 + t) * 768, nullptr, 0, 0,
                 hkqv[t], 768, MP, 768, 256);
        // relation-specific per-head transforms of k and v
        for (int e = 0; e < 5; ++e) {
            int s = srcOf[e];
            for (int h = 0; h < 4; ++h) {
                size_t rofs = (size_t)((l * 5 + e) * 4 + h) * 64 * 64;
                gemm(hkqv[s] + h * 64, 768, wT_krel + rofs, 64, nullptr,
                     nullptr, 0, 0, hke[e] + h * 64, HID, MP, 64, 64);
                gemm(hkqv[s] + 512 + h * 64, 768, wT_vrel + rofs, 64, nullptr,
                     nullptr, 0, 0, hve[e] + h * 64, HID, MP, 64, 64);
            }
        }
        // attention: segment softmax + aggregate
        fill(m_user, -3.0e38f, (long long)NNODE * 4);
        fill(s_user, 0.f, (long long)NNODE * 4);
        fill(agg_user, 0.f, (long long)NNODE * HID);
        const float* prl = p_rel + l * 20;
        k_att_assoc_logits<<<nb((long long)NNODE * 4), TPB, 0, stream>>>(
            hkqv[1], hke[0], hke[1], hke[2], prl, scale, la0, la1, la2, m_item);
        k_att_buys_logits<<<nb((long long)EBUYS * 4), TPB, 0, stream>>>(
            bsrc, bdst, hkqv[1], hkqv[0], hke[3], hke[4], prl, scale, lb3, lb4, m_item, m_user);
        k_att_assoc_exp<<<nb((long long)NNODE * 4), TPB, 0, stream>>>(la0, la1, la2, m_item, s_item);
        k_att_buys_exp<<<nb((long long)EBUYS * 4), TPB, 0, stream>>>(
            lb3, lb4, bsrc, bdst, m_item, m_user, s_item, s_user);
        k_att_assoc_agg<<<nb((long long)NNODE * HID), TPB, 0, stream>>>(
            la0, la1, la2, s_item, hve[0], hve[1], hve[2], agg_item);
        k_att_buys_agg<<<nb((long long)EBUYS * HID), TPB, 0, stream>>>(
            lb3, lb4, bsrc, bdst, s_item, s_user, hve[3], hve[4], agg_item, agg_user);
        // per-dst-type output projection + skip + residual + BN + relu
        for (int d = 0; d < 2; ++d) {
            const float* agg = (d == 0) ? agg_user : agg_item;
            k_gelu_f16<<<nb((long long)MP * HID), TPB, 0, stream>>>(agg, h_gelu, NNODE, MP);
            gemm(h_gelu, 256, wT_out + (size_t)(l * 5 + d) * 256 * 256, 256,
                 out_b + (size_t)(l * 5 + d) * 256, o_tmp, HID, NNODE, nullptr, 0, MP, 256, 256);
            k_fuse<<<nb(sect), TPB, 0, stream>>>(o_tmp, dout + d * sect, skip, l * 5 + d, sect);
            bn(o_tmp, fbn_g + (size_t)(l * 2 + d) * 256, fbn_b + (size_t)(l * 2 + d) * 256,
               dout + d * sect, hx[d]);
        }
    }
}
